// EdgeConv_25314537242840
// MI455X (gfx1250) — compile-verified, matrix operand order checked by reading, not run on hardware
//
#include <hip/hip_runtime.h>
#include <hip/hip_bf16.h>
#include <math.h>

typedef __attribute__((ext_vector_type(16))) _Float16 v16h;
typedef __attribute__((ext_vector_type(8)))  float    v8f;

#define NPTS   2048
#define NBATCH 8
#define NROWS  (NBATCH * NPTS)   // 16384
#define KNN_K  32

__device__ __forceinline__ v8f wmma32(v16h a, v16h b, v8f c) {
    // D(16x16,f32) = A(16x32,f16) x B(32x16,f16) + C
    return __builtin_amdgcn_wmma_f32_16x16x32_f16(
        /*neg_a=*/false, a, /*neg_b=*/false, b,
        /*c_mod=*/(short)0, c, /*reuse_a=*/false, /*reuse_b=*/false);
}

// ---------------------------------------------------------------------------
// 0) zero the BN-stats region (must be re-zeroed every call: harness doesn't)
// ---------------------------------------------------------------------------
__global__ void zero_kernel(float* __restrict__ p, int n) {
    int i = blockIdx.x * blockDim.x + threadIdx.x;
    if (i < n) p[i] = 0.0f;
}

// ---------------------------------------------------------------------------
// 1) convert weights f32 -> f16, folding the edge-feature subtraction into
//    the "conv A" matrices:  [nbr-ctr, ctr] @ [W1,W2]^T == [nbr, ctr] @ [W1, W2-W1]^T
//    layout in dst (halves): We2[0..4096) Wl1a'[..20480) Wl1b[..36864)
//                            Wl2a'[..69632) Wl2b[..86016)
// ---------------------------------------------------------------------------
__global__ void cvt_w_kernel(const float* __restrict__ We2,
                             const float* __restrict__ Wl1a,
                             const float* __restrict__ Wl1b,
                             const float* __restrict__ Wl2a,
                             const float* __restrict__ Wl2b,
                             _Float16* __restrict__ dst) {
    int i = blockIdx.x * 256 + threadIdx.x;
    float v;
    if (i < 4096) {
        v = We2[i];
    } else if (i < 20480) {                    // Wl1a: 128x128, CIN=64
        int j = i - 4096, c = j & 127;
        v = Wl1a[j];
        if (c >= 64) v -= Wl1a[j - 64];
    } else if (i < 36864) {
        v = Wl1b[i - 20480];
    } else if (i < 69632) {                    // Wl2a: 128x256, CIN=128
        int j = i - 36864, c = j & 255;
        v = Wl2a[j];
        if (c >= 128) v -= Wl2a[j - 128];
    } else if (i < 86016) {
        v = Wl2b[i - 69632];
    } else return;
    dst[i] = (_Float16)v;
}

// ---------------------------------------------------------------------------
// 2) KNN: one wave per point row, distances staged in LDS, 32 x wave-argmin
// ---------------------------------------------------------------------------
__global__ void knn_kernel(const float* __restrict__ x, int* __restrict__ idx) {
    __shared__ float dist[4][NPTS];            // 32 KB
    const int lane = threadIdx.x & 31;
    const int wave = threadIdx.x >> 5;
    const int row  = blockIdx.x * 4 + wave;    // 0..16383
    const int b = row >> 11, n = row & (NPTS - 1);
    const float* xb = x + (size_t)b * NPTS * 3;
    const float cx = xb[n * 3 + 0], cy = xb[n * 3 + 1], cz = xb[n * 3 + 2];
    for (int m = lane; m < NPTS; m += 32) {
        float dx = xb[m * 3 + 0] - cx;
        float dy = xb[m * 3 + 1] - cy;
        float dz = xb[m * 3 + 2] - cz;
        dist[wave][m] = dx * dx + dy * dy + dz * dz;
    }
    __syncthreads();
    int* out = idx + (size_t)row * KNN_K;
    for (int it = 0; it < KNN_K; ++it) {
        float best = 3.4e38f; int bpos = NPTS;
        for (int m = lane; m < NPTS; m += 32) {
            float d = dist[wave][m];
            if (d < best || (d == best && m < bpos)) { best = d; bpos = m; }
        }
        #pragma unroll
        for (int s = 16; s > 0; s >>= 1) {
            float ov = __shfl_xor(best, s, 32);
            int   op = __shfl_xor(bpos, s, 32);
            if (ov < best || (ov == best && op < bpos)) { best = ov; bpos = op; }
        }
        if (lane == 0) { out[it] = bpos; dist[wave][bpos] = 3.4e38f; }
        __syncthreads();
    }
}

// ---------------------------------------------------------------------------
// 3) emb1: z1 = x @ W_e1^T (3->64) + BN-stats accumulation
// ---------------------------------------------------------------------------
__global__ void e1_kernel(const float* __restrict__ x, const float* __restrict__ W,
                          float* __restrict__ z1, float* __restrict__ stats) {
    const int t = threadIdx.x;
    const int d = t & 63;
    const int row = blockIdx.x * 4 + (t >> 6);
    const float* xp = x + (size_t)row * 3;
    float z = xp[0] * W[d * 3 + 0] + xp[1] * W[d * 3 + 1] + xp[2] * W[d * 3 + 2];
    z1[(size_t)row * 64 + d] = z;
    __shared__ float s[64], ss[64];
    if (t < 64) { s[t] = 0.0f; ss[t] = 0.0f; }
    __syncthreads();
    atomicAdd(&s[d], z);
    atomicAdd(&ss[d], z * z);
    __syncthreads();
    if (t < 64) {
        atomicAdd(&stats[t], s[t]);
        atomicAdd(&stats[64 + t], ss[t]);
    }
}

// ---------------------------------------------------------------------------
// 4) finalize BN: sb[d] = gamma*rsqrt(var+eps); sb[C+d] = beta - mean*scale
// ---------------------------------------------------------------------------
__global__ void finalize_kernel(const float* __restrict__ stats,
                                const float* __restrict__ gamma,
                                const float* __restrict__ beta,
                                float* __restrict__ sb, int C, float inv_count) {
    int d = threadIdx.x;
    if (d < C) {
        float mean = stats[d] * inv_count;
        float var  = stats[C + d] * inv_count - mean * mean;
        float sc   = gamma[d] * rsqrtf(var + 1e-5f);
        sb[d]      = sc;
        sb[C + d]  = beta[d] - mean * sc;
    }
}

// ---------------------------------------------------------------------------
// 5) apply BN + ReLU, emit f16 feature tensor
// ---------------------------------------------------------------------------
__global__ void bnrelu_f16_kernel(const float* __restrict__ z, const float* __restrict__ sb,
                                  _Float16* __restrict__ h, int cmask, size_t total) {
    size_t i = (size_t)blockIdx.x * blockDim.x + threadIdx.x;
    if (i >= total) return;
    int d = (int)(i & (size_t)cmask);
    int C = cmask + 1;
    float v = z[i] * sb[d] + sb[C + d];
    h[i] = (_Float16)fmaxf(v, 0.0f);
}

// ---------------------------------------------------------------------------
// 6) emb2 via WMMA: z2(16384x64) = h1(16384x64,f16) @ We2(64x64,f16)^T + stats
// ---------------------------------------------------------------------------
__global__ void emb2_wmma_kernel(const _Float16* __restrict__ h1,
                                 const _Float16* __restrict__ W,
                                 float* __restrict__ z2, float* __restrict__ stats) {
    const int lane  = threadIdx.x & 31;
    const int wtile = blockIdx.x * (blockDim.x >> 5) + (threadIdx.x >> 5);
    const int row0  = wtile * 16;
    const int lrow  = lane & 15, lhi = lane >> 4;
    for (int nt = 0; nt < 4; ++nt) {
        v8f acc = {};
        #pragma unroll
        for (int kc = 0; kc < 2; ++kc) {
            v16h a  = *(const v16h*)(h1 + (size_t)(row0 + lrow) * 64 + kc * 32 + lhi * 16);
            v16h bm = *(const v16h*)(W  + (size_t)(nt * 16 + lrow) * 64 + kc * 32 + lhi * 16);
            acc = wmma32(a, bm, acc);
        }
        const int ch = nt * 16 + lrow;
        float s = 0.0f, ssq = 0.0f;
        #pragma unroll
        for (int i = 0; i < 8; ++i) {
            float v = acc[i];
            int r = row0 + i + (lhi << 3);
            z2[(size_t)r * 64 + ch] = v;
            s += v; ssq += v * v;
        }
        atomicAdd(&stats[ch], s);
        atomicAdd(&stats[64 + ch], ssq);
    }
}

// ---------------------------------------------------------------------------
// 7) EdgeConv local_op, one wave/point. Edge rows are now PURE GATHERS
//    (subtraction folded into Wa), staged via global_load_async_to_lds_b128.
//    PASS 1: stats of zA = edge(32x2C) @ Wa'(128x2C)^T
//    PASS 2: hA = relu(bn(zA)); stats of zB = hA @ Wb^T
//    PASS 3: hB = relu(bn(zB)); out = max over k (32 rows)
// ---------------------------------------------------------------------------
template <int CIN, int PASS, bool OUT16>
__global__ void local_kernel(const _Float16* __restrict__ hin,
                             const int* __restrict__ idx,
                             const _Float16* __restrict__ Wa,
                             const _Float16* __restrict__ Wb,
                             const float* __restrict__ sbA,
                             const float* __restrict__ sbB,
                             float* __restrict__ statsOut,
                             _Float16* __restrict__ out16,
                             float* __restrict__ out32) {
    constexpr int CE  = 2 * CIN;     // edge-feature width (128 or 256)
    constexpr int KCA = CE / 32;     // K-chunks for conv-A
    constexpr int EP  = CE + 16;     // +32B pad: avoids bank conflicts, keeps 32B align
    constexpr int HP  = 128 + 16;
    __shared__ __align__(32) _Float16 e [2][32][EP];
    __shared__ __align__(32) _Float16 ha[2][32][HP];

    const int lane = threadIdx.x & 31;
    const int p    = threadIdx.x >> 5;            // point slot in block (0..1)
    const int pt   = blockIdx.x * 2 + p;          // 0..16383
    const int b    = pt >> 11;

    // ---- gather via async DMA: lane k fills edge row k = [nbr_k, ctr] ----
    {
        const int nb = idx[(size_t)pt * KNN_K + lane];
        const _Float16* nbrp = hin + ((size_t)(b << 11) + nb) * CIN;
        // bias center pointer by -CIN halves: instr offset is added to BOTH
        // the LDS and global address, so offset 2*(CIN+c) hits ctr[c].
        const _Float16* ctrp = hin + (size_t)pt * CIN - CIN;
        unsigned eoff = (unsigned)(uintptr_t)(&e[p][lane][0]);  // LDS byte offset
        #pragma unroll
        for (int c = 0; c < CIN; c += 8)
            asm volatile("global_load_async_to_lds_b128 %0, %1, off offset:%2"
                         :: "v"(eoff), "v"(nbrp), "i"(2 * c) : "memory");
        #pragma unroll
        for (int c = 0; c < CIN; c += 8)
            asm volatile("global_load_async_to_lds_b128 %0, %1, off offset:%2"
                         :: "v"(eoff), "v"(ctrp), "i"(2 * (CIN + c)) : "memory");
        asm volatile("s_wait_asynccnt 0x0" ::: "memory");
    }
    __syncthreads();

    const int lrow = lane & 15, lhi = lane >> 4;

    // ---- conv A: zA(32x128) = e(32xCE) @ Wa'(128xCE)^T ----
    for (int nt = 0; nt < 8; ++nt) {
        const int ch = nt * 16 + lrow;
        v16h bf[KCA];
        #pragma unroll
        for (int kc = 0; kc < KCA; ++kc)
            bf[kc] = *(const v16h*)(Wa + (size_t)ch * CE + kc * 32 + lhi * 16);
        float s = 0.0f, ssq = 0.0f;
        #pragma unroll
        for (int mt = 0; mt < 2; ++mt) {
            v8f acc = {};
            #pragma unroll
            for (int kc = 0; kc < KCA; ++kc) {
                v16h a = *(const v16h*)(&e[p][mt * 16 + lrow][kc * 32 + lhi * 16]);
                acc = wmma32(a, bf[kc], acc);
            }
            if (PASS == 1) {
                #pragma unroll
                for (int i = 0; i < 8; ++i) { float v = acc[i]; s += v; ssq += v * v; }
            } else {
                const float sc = sbA[ch], bi = sbA[128 + ch];
                #pragma unroll
                for (int i = 0; i < 8; ++i) {
                    float v = fmaxf(acc[i] * sc + bi, 0.0f);
                    ha[p][mt * 16 + i + (lhi << 3)][ch] = (_Float16)v;
                }
            }
        }
        if (PASS == 1) {
            atomicAdd(&statsOut[ch], s);
            atomicAdd(&statsOut[128 + ch], ssq);
        }
    }
    if (PASS == 1) return;
    __syncthreads();

    // ---- conv B: zB(32x128) = hA(32x128) @ Wb(128x128)^T ----
    for (int nt = 0; nt < 8; ++nt) {
        const int ch = nt * 16 + lrow;
        v16h bf[4];
        #pragma unroll
        for (int kc = 0; kc < 4; ++kc)
            bf[kc] = *(const v16h*)(Wb + (size_t)ch * 128 + kc * 32 + lhi * 16);
        float s = 0.0f, ssq = 0.0f;
        float mx = -3.4e38f;
        #pragma unroll
        for (int mt = 0; mt < 2; ++mt) {
            v8f acc = {};
            #pragma unroll
            for (int kc = 0; kc < 4; ++kc) {
                v16h a = *(const v16h*)(&ha[p][mt * 16 + lrow][kc * 32 + lhi * 16]);
                acc = wmma32(a, bf[kc], acc);
            }
            if (PASS == 2) {
                #pragma unroll
                for (int i = 0; i < 8; ++i) { float v = acc[i]; s += v; ssq += v * v; }
            } else {
                const float sc = sbB[ch], bi = sbB[128 + ch];
                #pragma unroll
                for (int i = 0; i < 8; ++i)
                    mx = fmaxf(mx, fmaxf(acc[i] * sc + bi, 0.0f));
            }
        }
        if (PASS == 2) {
            atomicAdd(&statsOut[ch], s);
            atomicAdd(&statsOut[128 + ch], ssq);
        } else {
            mx = fmaxf(mx, __shfl_xor(mx, 16, 32));   // merge the two row-halves
            if (lane < 16) {
                if (OUT16) out16[(size_t)pt * 128 + ch] = (_Float16)mx;
                else       out32[(size_t)pt * 128 + ch] = mx;
            }
        }
    }
}

// ---------------------------------------------------------------------------
// host-side launcher
// ---------------------------------------------------------------------------
extern "C" void kernel_launch(void* const* d_in, const int* in_sizes, int n_in,
                              void* d_out, int out_size, void* d_ws, size_t ws_size,
                              hipStream_t stream) {
    const float* x     = (const float*)d_in[0];
    const float* W_e1  = (const float*)d_in[1];
    const float* g_e1  = (const float*)d_in[2];
    const float* b_e1  = (const float*)d_in[3];
    const float* W_e2  = (const float*)d_in[4];
    const float* g_e2  = (const float*)d_in[5];
    const float* b_e2  = (const float*)d_in[6];
    const float* W_l1a = (const float*)d_in[7];
    const float* g_l1a = (const float*)d_in[8];
    const float* b_l1a = (const float*)d_in[9];
    const float* W_l1b = (const float*)d_in[10];
    const float* g_l1b = (const float*)d_in[11];
    const float* b_l1b = (const float*)d_in[12];
    const float* W_l2a = (const float*)d_in[13];
    const float* g_l2a = (const float*)d_in[14];
    const float* b_l2a = (const float*)d_in[15];
    const float* W_l2b = (const float*)d_in[16];
    const float* g_l2b = (const float*)d_in[17];
    const float* b_l2b = (const float*)d_in[18];
    (void)in_sizes; (void)n_in; (void)out_size; (void)ws_size;

    char* ws = (char*)d_ws;
    int*      idx    = (int*)      (ws + 0);            //  2 MB: [16384][32] int
    float*    z1     = (float*)    (ws + (2u  << 20));  //  4 MB: [16384][64]
    float*    z2     = (float*)    (ws + (6u  << 20));  //  4 MB: [16384][64]
    _Float16* h1f16  = (_Float16*) (ws + (10u << 20));  //  2 MB
    _Float16* hfeat  = (_Float16*) (ws + (12u << 20));  //  2 MB: local1 input
    _Float16* h2feat = (_Float16*) (ws + (14u << 20));  //  4 MB: local2 input
    _Float16* wf16   = (_Float16*) (ws + (18u << 20));  //  172 KB
    float*    stats  = (float*)    (ws + (18u << 20) + 200 * 1024);  // 1280 f32
    float*    sb     = (float*)    (ws + (18u << 20) + 208 * 1024);  // 1280 f32

    _Float16* wf_e2  = wf16 + 0;
    _Float16* wf_l1a = wf16 + 4096;
    _Float16* wf_l1b = wf16 + 20480;
    _Float16* wf_l2a = wf16 + 36864;
    _Float16* wf_l2b = wf16 + 69632;

    float* st_e1  = stats + 0;    float* sb_e1  = sb + 0;     // 64 ch
    float* st_e2  = stats + 128;  float* sb_e2  = sb + 128;   // 64 ch
    float* st_l1a = stats + 256;  float* sb_l1a = sb + 256;   // 128 ch
    float* st_l1b = stats + 512;  float* sb_l1b = sb + 512;
    float* st_l2a = stats + 768;  float* sb_l2a = sb + 768;
    float* st_l2b = stats + 1024; float* sb_l2b = sb + 1024;

    const float invBN  = 1.0f / (float)NROWS;             // BN over (B,N)
    const float invBNK = 1.0f / (float)(NROWS * KNN_K);   // BN over (B,N,K)

    zero_kernel<<<(1280 + 255) / 256, 256, 0, stream>>>(stats, 1280);
    cvt_w_kernel<<<(86016 + 255) / 256, 256, 0, stream>>>(W_e2, W_l1a, W_l1b, W_l2a, W_l2b, wf16);
    knn_kernel<<<NROWS / 4, 128, 0, stream>>>(x, idx);

    // embedding stage
    e1_kernel<<<NROWS / 4, 256, 0, stream>>>(x, W_e1, z1, st_e1);
    finalize_kernel<<<1, 256, 0, stream>>>(st_e1, g_e1, b_e1, sb_e1, 64, invBN);
    bnrelu_f16_kernel<<<(NROWS * 64 + 255) / 256, 256, 0, stream>>>(z1, sb_e1, h1f16, 63, (size_t)NROWS * 64);
    emb2_wmma_kernel<<<NROWS / 32, 64, 0, stream>>>(h1f16, wf_e2, z2, st_e2);
    finalize_kernel<<<1, 256, 0, stream>>>(st_e2, g_e2, b_e2, sb_e2, 64, invBN);
    bnrelu_f16_kernel<<<(NROWS * 64 + 255) / 256, 256, 0, stream>>>(z2, sb_e2, hfeat, 63, (size_t)NROWS * 64);

    // EdgeConv block 1 (CIN=64): 3 passes for global BN
    local_kernel<64, 1, true><<<NROWS / 2, 64, 0, stream>>>(hfeat, idx, wf_l1a, wf_l1b, sb_l1a, sb_l1b, st_l1a, h2feat, nullptr);
    finalize_kernel<<<1, 256, 0, stream>>>(st_l1a, g_l1a, b_l1a, sb_l1a, 128, invBNK);
    local_kernel<64, 2, true><<<NROWS / 2, 64, 0, stream>>>(hfeat, idx, wf_l1a, wf_l1b, sb_l1a, sb_l1b, st_l1b, h2feat, nullptr);
    finalize_kernel<<<1, 256, 0, stream>>>(st_l1b, g_l1b, b_l1b, sb_l1b, 128, invBNK);
    local_kernel<64, 3, true><<<NROWS / 2, 64, 0, stream>>>(hfeat, idx, wf_l1a, wf_l1b, sb_l1a, sb_l1b, st_l1b, h2feat, nullptr);

    // EdgeConv block 2 (CIN=128), writes f32 output
    float* out = (float*)d_out;
    local_kernel<128, 1, false><<<NROWS / 2, 64, 0, stream>>>(h2feat, idx, wf_l2a, wf_l2b, sb_l2a, sb_l2b, st_l2a, nullptr, out);
    finalize_kernel<<<1, 256, 0, stream>>>(st_l2a, g_l2a, b_l2a, sb_l2a, 128, invBNK);
    local_kernel<128, 2, false><<<NROWS / 2, 64, 0, stream>>>(h2feat, idx, wf_l2a, wf_l2b, sb_l2a, sb_l2b, st_l2b, nullptr, out);
    finalize_kernel<<<1, 256, 0, stream>>>(st_l2b, g_l2b, b_l2b, sb_l2b, 128, invBNK);
    local_kernel<128, 3, false><<<NROWS / 2, 64, 0, stream>>>(h2feat, idx, wf_l2a, wf_l2b, sb_l2a, sb_l2b, st_l2b, nullptr, out);
}
